// YatSphericalCausalAttention_64793876627786
// MI455X (gfx1250) — matrix operand order repaired and checked by reading, not compiled, not run on hardware
//
#include <hip/hip_runtime.h>
#include <stdint.h>

// ---------------------------------------------------------------------------
// Yat spherical causal attention for MI455X (gfx1250, wave32, WMMA bf16)
//   B=2, L=2048, E=1024, H=16, D=64, C=2.01
// ---------------------------------------------------------------------------

#define EMBED 1024
#define NHEAD 16
#define HDIM  64
#define SEQ   2048
#define BATCH 2
#define CEPS  2.01f

typedef __attribute__((ext_vector_type(16))) __bf16 v16bf;
typedef __attribute__((ext_vector_type(8)))  float  v8f;

static __device__ __forceinline__ unsigned short f2bf(float f) {
  uint32_t u = __float_as_uint(f);
  uint32_t r = u + 0x7FFFu + ((u >> 16) & 1u);   // round-to-nearest-even
  return (unsigned short)(r >> 16);
}
static __device__ __forceinline__ float bf2f(unsigned short h) {
  return __uint_as_float(((uint32_t)h) << 16);
}
static __device__ __forceinline__ uint32_t pack2(float a, float b) {
  return (uint32_t)f2bf(a) | ((uint32_t)f2bf(b) << 16);
}

union Frag {
  uint32_t u[8];
  uint4    q2[2];
  v16bf    v;
};

// A-fragment layout (16x32 bf16): per-lane base p = row(lo)*ld + 16*hi;
// K[k0 .. +15] contiguous
#define LDA(f, p, k)                                   \
  {                                                    \
    (f).q2[0] = *(const uint4*)((p) + (k));            \
    (f).q2[1] = *(const uint4*)((p) + (k) + 8);        \
  }
// B-fragment layout (32x16 bf16 from B^T rows): per-lane base p = row(lo)*ld + 16*hi;
// K[k0 .. +15] contiguous
#define LDB(f, p, k)                                   \
  {                                                    \
    (f).q2[0] = *(const uint4*)((p) + (k));            \
    (f).q2[1] = *(const uint4*)((p) + (k) + 8);        \
  }

static __device__ __forceinline__ v8f wmma_bf16(const Frag& a, const Frag& b, v8f c) {
  return __builtin_amdgcn_wmma_f32_16x16x32_bf16(false, a.v, false, b.v,
                                                 (short)0, c, false, false);
}

// ---------------------------------------------------------------------------
// Stage 0: fp32 -> bf16 conversion / weight transpose (K x N) -> (N x K)
// ---------------------------------------------------------------------------
__global__ void cvt_bf16_kernel(const float* __restrict__ in,
                                unsigned short* __restrict__ out, int n) {
  int i = blockIdx.x * blockDim.x + threadIdx.x;
  if (i < n) out[i] = f2bf(in[i]);
}

__global__ void transpose_bf16_kernel(const float* __restrict__ in,
                                      unsigned short* __restrict__ out,
                                      int K, int N) {
  int i = blockIdx.x * blockDim.x + threadIdx.x;
  if (i >= K * N) return;
  int n = i / K;
  int k = i - n * K;
  out[(size_t)n * K + k] = f2bf(in[(size_t)k * N + n]);
}

// ---------------------------------------------------------------------------
// Stage 1/4: GEMM  C[M,N] = A[M,K](bf16) * B^T[N,K](bf16) + bias
//   wave = 32x64 tile, 8 accumulators, single-buffered fragments
//   (fragments declared in-loop so the allocator renames them; ~135 live VGPRs,
//    no spills; next iteration's loads issue as soon as current WMMAs issue)
// ---------------------------------------------------------------------------
__global__ void gemm_bf16_wmma_kernel(const unsigned short* __restrict__ A,
                                      const unsigned short* __restrict__ Bt,
                                      const float* __restrict__ bias,
                                      float* __restrict__ Cf,
                                      unsigned short* __restrict__ Cb,
                                      int M, int N, int K, int f32out) {
  int wid  = (int)((blockIdx.x * blockDim.x + threadIdx.x) >> 5);
  int lane = threadIdx.x & 31;
  int lo = lane & 15, hi = lane >> 4;
  int mTiles = M >> 5;                   // 32-row tiles
  int m0 = (wid % mTiles) << 5;
  int n0 = (wid / mTiles) << 6;
  if (n0 >= N) return;

  // per-lane row pointers (hoisted address math)
  const unsigned short* pA0 = A  + (size_t)(m0 + lo) * K + (hi << 4);
  const unsigned short* pA1 = pA0 + (size_t)16 * K;
  const unsigned short* pB0 = Bt + (size_t)(n0 + lo) * K + (hi << 4);
  const unsigned short* pB1 = pB0 + (size_t)16 * K;
  const unsigned short* pB2 = pB0 + (size_t)32 * K;
  const unsigned short* pB3 = pB0 + (size_t)48 * K;

  v8f acc[8] = {};   // acc[mi*4 + ni]

  for (int k0 = 0; k0 < K; k0 += 32) {
    __builtin_prefetch((const void*)(pA0 + k0 + 96), 0, 1);

    Frag a0, a1, b0, b1, b2, b3;
    LDA(a0, pA0, k0); LDA(a1, pA1, k0);
    LDB(b0, pB0, k0); LDB(b1, pB1, k0);
    LDB(b2, pB2, k0); LDB(b3, pB3, k0);

    acc[0] = wmma_bf16(a0, b0, acc[0]);
    acc[1] = wmma_bf16(a0, b1, acc[1]);
    acc[2] = wmma_bf16(a0, b2, acc[2]);
    acc[3] = wmma_bf16(a0, b3, acc[3]);
    acc[4] = wmma_bf16(a1, b0, acc[4]);
    acc[5] = wmma_bf16(a1, b1, acc[5]);
    acc[6] = wmma_bf16(a1, b2, acc[6]);
    acc[7] = wmma_bf16(a1, b3, acc[7]);
  }

#pragma unroll
  for (int mi = 0; mi < 2; mi++) {
#pragma unroll
    for (int ni = 0; ni < 4; ni++) {
      int col  = n0 + (ni << 4) + lo;
      float bv = bias[col];
#pragma unroll
      for (int r = 0; r < 8; r++) {
        int row    = m0 + (mi << 4) + (hi << 3) + r;   // D-layout: lanes 16-31 = M+8
        float val  = acc[mi * 4 + ni][r] + bv;
        size_t idx = (size_t)row * N + col;
        if (f32out) Cf[idx] = val;
        else        Cb[idx] = f2bf(val);
      }
    }
  }
}

// ---------------------------------------------------------------------------
// Stage 2: split heads, L2-normalize q/k, transpose v -> [B,H,D,L]
//   one wave per (b, l, h); lanes cover D=64 two elements each
// ---------------------------------------------------------------------------
__global__ void normheads_kernel(const unsigned short* __restrict__ qkv,
                                 unsigned short* __restrict__ qn,
                                 unsigned short* __restrict__ kn,
                                 unsigned short* __restrict__ vt) {
  int wid  = (int)((blockIdx.x * blockDim.x + threadIdx.x) >> 5);
  int lane = threadIdx.x & 31;
  int h = wid & (NHEAD - 1);
  int l = (wid >> 4) & (SEQ - 1);
  int b = wid >> 15;   // 4 + 11 bits

  size_t rowbase = (size_t)(b * SEQ + l) * (3 * EMBED);
  size_t hb      = ((size_t)((b * NHEAD + h) * SEQ) + l) * HDIM;

  // q
  {
    size_t off = rowbase + h * HDIM;
    float q0 = bf2f(qkv[off + lane]);
    float q1 = bf2f(qkv[off + 32 + lane]);
    float ss = q0 * q0 + q1 * q1;
    for (int m = 16; m >= 1; m >>= 1) ss += __shfl_xor(ss, m, 32);
    float rn = rsqrtf(ss + 1e-12f);
    qn[hb + lane]      = f2bf(q0 * rn);
    qn[hb + 32 + lane] = f2bf(q1 * rn);
  }
  // k
  {
    size_t off = rowbase + EMBED + h * HDIM;
    float k0 = bf2f(qkv[off + lane]);
    float k1 = bf2f(qkv[off + 32 + lane]);
    float ss = k0 * k0 + k1 * k1;
    for (int m = 16; m >= 1; m >>= 1) ss += __shfl_xor(ss, m, 32);
    float rn = rsqrtf(ss + 1e-12f);
    kn[hb + lane]      = f2bf(k0 * rn);
    kn[hb + 32 + lane] = f2bf(k1 * rn);
  }
  // v -> transposed [B,H,D,L]
  {
    size_t off = rowbase + 2 * EMBED + h * HDIM;
    size_t vb  = (size_t)(b * NHEAD + h) * HDIM * SEQ;
    vt[vb + (size_t)lane * SEQ + l]        = qkv[off + lane];
    vt[vb + (size_t)(lane + 32) * SEQ + l] = qkv[off + 32 + lane];
  }
}

// ---------------------------------------------------------------------------
// Stage 3: flash causal Yat attention. One wave = 16 queries of one (b,h).
//   S^T = Kn * Qn^T (D-layout: lane = query, VGPR = key)
//   -> score/softmax per-lane (+1 shfl_xor(16) combine)
//   -> P^T packed into B-fragment via half-wave exchange
//   -> O^T += V^T * P^T   (lane = query, VGPR = d)
// ---------------------------------------------------------------------------
__global__ void attn_kernel(const unsigned short* __restrict__ qn,
                            const unsigned short* __restrict__ kn,
                            const unsigned short* __restrict__ vt,
                            unsigned short* __restrict__ ao) {
  int wid  = (int)((blockIdx.x * blockDim.x + threadIdx.x) >> 5);
  int lane = threadIdx.x & 31;
  int lo = lane & 15, hi = lane >> 4;

  int qb = wid & 127;          // SEQ/16 = 128 query blocks
  int h  = (wid >> 7) & (NHEAD - 1);
  int b  = wid >> 11;
  int q0 = qb << 4;

  const unsigned short* qn_bh = qn + (size_t)(b * NHEAD + h) * SEQ * HDIM;
  const unsigned short* kn_bh = kn + (size_t)(b * NHEAD + h) * SEQ * HDIM;
  const unsigned short* vt_bh = vt + (size_t)(b * NHEAD + h) * HDIM * SEQ;

  // per-lane base pointers
  const unsigned short* pQ = qn_bh + (size_t)(q0 + lo) * HDIM + (hi << 4);  // B-frag base
  const unsigned short* pK = kn_bh + (size_t)lo * HDIM + (hi << 4);         // A-frag base
  const unsigned short* pV = vt_bh + (size_t)lo * SEQ + (hi << 4);          // A-frag base

  Frag qf0, qf1;                       // Qn^T as B-fragments, reused all blocks
  LDB(qf0, pQ, 0);
  LDB(qf1, pQ, 32);

  v8f o[4] = {};
  float m_run = -1e30f, l_run = 0.0f;
  int   qidx  = q0 + lo;

  for (int k0 = 0; k0 <= q0 + 15; k0 += 32) {
    // ---- load all K fragments (2 key sub-tiles x 2 d-chunks) and V fragments up front
    Frag kf00, kf01, kf10, kf11;
    const unsigned short* pK0 = pK + (size_t)k0 * HDIM;
    const unsigned short* pK1 = pK0 + (size_t)16 * HDIM;
    LDA(kf00, pK0, 0); LDA(kf01, pK0, 32);
    LDA(kf10, pK1, 0); LDA(kf11, pK1, 32);
    Frag vf0, vf1, vf2, vf3;
    LDA(vf0, pV, k0);
    LDA(vf1, pV + (size_t)16 * SEQ, k0);
    LDA(vf2, pV + (size_t)32 * SEQ, k0);
    LDA(vf3, pV + (size_t)48 * SEQ, k0);

    // ---- S^T tiles: tile0 = keys k0..k0+15, tile1 = keys k0+16..k0+31
    v8f s0 = {}, s1 = {};
    s0 = wmma_bf16(kf00, qf0, s0);
    s0 = wmma_bf16(kf01, qf1, s0);
    s1 = wmma_bf16(kf10, qf0, s1);
    s1 = wmma_bf16(kf11, qf1, s1);

    // ---- Yat score + causal mask (key index depends on VGPR & lane half)
    float sc0[8], sc1[8];
#pragma unroll
    for (int r = 0; r < 8; r++) {
      int key0 = k0 + (hi << 3) + r;
      int key1 = k0 + 16 + (hi << 3) + r;
      float x0 = fminf(fmaxf(s0[r], -1.0f), 1.0f);
      float x1 = fminf(fmaxf(s1[r], -1.0f), 1.0f);
      sc0[r] = (key0 <= qidx) ? (x0 * x0) / (CEPS - 2.0f * x0) : -1e9f;
      sc1[r] = (key1 <= qidx) ? (x1 * x1) / (CEPS - 2.0f * x1) : -1e9f;
    }

    // ---- online softmax (per query; lane & lane^16 share a query)
    float mloc = -1e30f;
#pragma unroll
    for (int r = 0; r < 8; r++) mloc = fmaxf(mloc, fmaxf(sc0[r], sc1[r]));
    mloc = fmaxf(mloc, __shfl_xor(mloc, 16, 32));
    float m_new = fmaxf(m_run, mloc);
    float scale = __expf(m_run - m_new);

    float p0[8], p1[8], lsum = 0.0f;
#pragma unroll
    for (int r = 0; r < 8; r++) {
      p0[r] = __expf(sc0[r] - m_new);
      p1[r] = __expf(sc1[r] - m_new);
      lsum += p0[r] + p1[r];
    }
    lsum += __shfl_xor(lsum, 16, 32);
    l_run = l_run * scale + lsum;
#pragma unroll
    for (int t = 0; t < 4; t++)
#pragma unroll
      for (int j = 0; j < 8; j++) o[t][j] *= scale;

    // ---- P^T D-layout -> B-fragment: half-wave exchange via shfl_xor(16)
    Frag pf;
#pragma unroll
    for (int i = 0; i < 4; i++) {
      uint32_t t0 = pack2(p0[2 * i], p0[2 * i + 1]);
      uint32_t t1 = pack2(p1[2 * i], p1[2 * i + 1]);
      uint32_t x0 = __shfl_xor(t0, 16, 32);
      uint32_t x1 = __shfl_xor(t1, 16, 32);
      pf.u[i]     = hi ? x1 : t0;   // elems 0..7 : keys {0..7} | {16..23}
      pf.u[4 + i] = hi ? t1 : x0;   // elems 8..15: keys {8..15}| {24..31}
    }

    // ---- O^T += V^T * P^T  (4 d-tiles of 16)
    o[0] = wmma_bf16(vf0, pf, o[0]);
    o[1] = wmma_bf16(vf1, pf, o[1]);
    o[2] = wmma_bf16(vf2, pf, o[2]);
    o[3] = wmma_bf16(vf3, pf, o[3]);

    m_run = m_new;
  }

  // ---- normalize & store: per lane, 8 consecutive d per tile -> one b128 store
  float inv = 1.0f / l_run;
  size_t rowbase = (size_t)(b * SEQ + q0 + lo) * EMBED + h * HDIM;
#pragma unroll
  for (int t = 0; t < 4; t++) {
    uint4 st;
    st.x = pack2(o[t][0] * inv, o[t][1] * inv);
    st.y = pack2(o[t][2] * inv, o[t][3] * inv);
    st.z = pack2(o[t][4] * inv, o[t][5] * inv);
    st.w = pack2(o[t][6] * inv, o[t][7] * inv);
    *(uint4*)(ao + rowbase + (t << 4) + (hi << 3)) = st;
  }
}

// ---------------------------------------------------------------------------
// Host launcher
// ---------------------------------------------------------------------------
extern "C" void kernel_launch(void* const* d_in, const int* in_sizes, int n_in,
                              void* d_out, int out_size, void* d_ws, size_t ws_size,
                              hipStream_t stream) {
  const float* x    = (const float*)d_in[0];  // [2,2048,1024]
  const float* Wqkv = (const float*)d_in[1];  // [1024,3072]
  const float* bqkv = (const float*)d_in[2];  // [3072]
  const float* Wout = (const float*)d_in[3];  // [1024,1024]
  const float* bout = (const float*)d_in[4];  // [1024]
  float* out = (float*)d_out;
  unsigned short* ws = (unsigned short*)d_ws;

  const size_t N_X   = (size_t)BATCH * SEQ * EMBED;          // 4,194,304
  const size_t N_WQ  = (size_t)3 * EMBED * EMBED;            // 3,145,728
  const size_t N_WO  = (size_t)EMBED * EMBED;                // 1,048,576
  const size_t N_QKV = (size_t)BATCH * SEQ * 3 * EMBED;      // 12,582,912
  const size_t N_H   = (size_t)BATCH * NHEAD * SEQ * HDIM;   // 4,194,304

  const size_t XB  = 0;
  const size_t WQT = XB + N_X;
  const size_t WOT = WQT + N_WQ;
  const size_t QKV = WOT + N_WO;
  const size_t QN  = QKV + N_QKV;
  const size_t KN  = QN + N_H;
  const size_t VT  = KN + N_H;
  const size_t AO  = VT + N_H;

  // Stage 0: bf16 staging
  cvt_bf16_kernel<<<(int)((N_X + 255) / 256), 256, 0, stream>>>(x, ws + XB, (int)N_X);
  transpose_bf16_kernel<<<(int)((N_WQ + 255) / 256), 256, 0, stream>>>(Wqkv, ws + WQT, EMBED, 3 * EMBED);
  transpose_bf16_kernel<<<(int)((N_WO + 255) / 256), 256, 0, stream>>>(Wout, ws + WOT, EMBED, EMBED);

  // Stage 1: qkv = x @ Wqkv + b   (M=4096, N=3072, K=1024) -> bf16
  // tiles = (4096/32)*(3072/64) = 6144 waves; 4 waves/block
  gemm_bf16_wmma_kernel<<<6144 / 4, 128, 0, stream>>>(
      ws + XB, ws + WQT, bqkv, nullptr, ws + QKV, BATCH * SEQ, 3 * EMBED, EMBED, 0);

  // Stage 2: heads + normalize + v-transpose (one wave per (b,l,h))
  normheads_kernel<<<(BATCH * SEQ * NHEAD) / 4, 128, 0, stream>>>(
      ws + QKV, ws + QN, ws + KN, ws + VT);

  // Stage 3: flash Yat attention (one wave per 16 queries): 2*16*128 = 4096 waves
  attn_kernel<<<4096 / 4, 128, 0, stream>>>(ws + QN, ws + KN, ws + VT, ws + AO);

  // Stage 4: out = ao @ Wout + b  (M=4096, N=1024, K=1024) -> fp32 d_out
  // tiles = (4096/32)*(1024/64) = 2048 waves
  gemm_bf16_wmma_kernel<<<2048 / 4, 128, 0, stream>>>(
      ws + AO, ws + WOT, bout, out, nullptr, BATCH * SEQ, EMBED, EMBED, 1);
}
